// StyleEncoder_67860483276833
// MI455X (gfx1250) — matrix operand order repaired and checked by reading, not compiled, not run on hardware
//
#include <hip/hip_runtime.h>
#include <hip/hip_bf16.h>
#include <cstdint>
#include <cstddef>

// ---------------------------------------------------------------------------
// StyleEncoder for MI455X (gfx1250, wave32, WMMA).
// All dense matmuls run through V_WMMA_F32_16X16X32_BF16 (bf16 in, f32 acc).
// GEMM tiles are staged global->LDS with GLOBAL_LOAD_ASYNC_TO_LDS_B128
// (ASYNCcnt) into double-buffered LDS so the DMA overlaps the WMMA pipe;
// one barrier per K-step. Activations are kept as bf16 shadows so the hot
// loops do zero format conversion. f32 masters only where residual/LN needs.
// Dims: DIM=256, HEADS=8, HD=32, DFF=128, DEPTH=2, T=64, 256 segs, L=8.
// Spatial rows Ms = 16384*9 = 147456 (mult of 64). Temporal rows Mt = 16384.
// ---------------------------------------------------------------------------

typedef __attribute__((ext_vector_type(16))) __bf16 v16bf;
typedef __attribute__((ext_vector_type(8)))  float  v8f;

union Frag16 {
  v16bf v;
  uint4 q[2];
  unsigned short h[16];
};

__device__ __forceinline__ unsigned short f2bf(float f) {
  unsigned u = __float_as_uint(f);
  unsigned r = u + 0x7FFFu + ((u >> 16) & 1u);   // round-to-nearest-even
  return (unsigned short)(r >> 16);
}

__device__ __forceinline__ v8f wmma_bf16(v16bf a, v16bf b, v8f c) {
  // D = A(16x32 bf16) x B(32x16 bf16) + C(16x16 f32)
  return __builtin_amdgcn_wmma_f32_16x16x32_bf16(
      /*neg_a=*/false, a, /*neg_b=*/false, b,
      /*c_mod=*/(short)0, c, /*reuse_a=*/false, /*reuse_b=*/false);
}

// --------------------------- weight f32 -> bf16 ----------------------------
__global__ void cvt_kernel(const float* __restrict__ src,
                           unsigned short* __restrict__ dst, int n) {
  int i = blockIdx.x * 256 + threadIdx.x;
  if (i < n) dst[i] = f2bf(src[i]);
}

// ------------------------ input embed + GELU + CLS -------------------------
// X[(t*256+seg)*9 + tok, d]; tok 0 is the CLS token (1e-5).
__global__ __launch_bounds__(256) void embed_kernel(
    const float* __restrict__ mv, const int* __restrict__ sse,
    const float* __restrict__ w_in, const float* __restrict__ b_in,
    float* __restrict__ X, unsigned short* __restrict__ Xh) {
  const int row = blockIdx.x;          // 0 .. 147455
  const int d   = threadIdx.x;         // 0 .. 255
  const int tok = row % 9;
  const int ts  = row / 9;             // t*256 + seg
  float val;
  if (tok == 0) {
    val = 1e-5f;
  } else {
    const int t   = ts >> 8;
    const int seg = ts & 255;
    const int n   = sse[seg * 2] + (tok - 1);
    const float m0 = mv[((size_t)n * 64 + t) * 2 + 0];
    const float m1 = mv[((size_t)n * 64 + t) * 2 + 1];
    const float z  = m0 * w_in[d * 2] + m1 * w_in[d * 2 + 1] + b_in[d];
    val = 0.5f * z * (1.0f + erff(z * 0.70710678118654752f)); // exact GELU
  }
  X[(size_t)row * 256 + d]  = val;
  Xh[(size_t)row * 256 + d] = f2bf(val);
}

// ------------------------------- WMMA GEMM ---------------------------------
// C[M,N] = act(A[M,K] @ W[N,K]^T + bias), A/W bf16, C -> f32 and/or bf16.
// M mult of 64, N mult of 128, K mult of 32. Block: 256 threads (8 waves),
// 64x128 tile; each wave owns one 16-row strip x four 16-col strips
// (1 A-frag reused across 4 WMMAs per K-step). Double-buffered LDS fed by
// async global->LDS DMA; one barrier per K-step.
__global__ __launch_bounds__(256) void gemm_bf16(
    const unsigned short* __restrict__ A, const unsigned short* __restrict__ W,
    const float* __restrict__ bias, float* __restrict__ Cf,
    unsigned short* __restrict__ Ch, int M, int N, int K, int relu) {
  __shared__ unsigned short As[2][64 * 32];    // 2 x 4 KB
  __shared__ unsigned short Bs[2][128 * 32];   // 2 x 8 KB
  const int tid  = threadIdx.x;
  const int lane = tid & 31;
  const int wave = tid >> 5;
  const int ln15 = lane & 15;
  const int hi   = lane >> 4;
  const int blockM = blockIdx.x * 64;
  const int blockN = blockIdx.y * 128;
  const int wm  = wave >> 1;            // 0..3 row strip
  const int wn0 = (wave & 1) * 4;       // 0 or 4: first of 4 col strips
  const int lr  = tid >> 2;             // staging row 0..63
  const int lc  = (tid & 3) * 8;        // staging col base 0/8/16/24
  const int sOff = lr * 32 + lc;

  const unsigned short* gA  = A + (size_t)(blockM + lr) * K + lc;
  const unsigned short* gB0 = W + (size_t)(blockN + lr) * K + lc;
  const unsigned short* gB1 = W + (size_t)(blockN + lr + 64) * K + lc;

#if defined(__gfx1250__)
  // Async DMA global->LDS (16B per lane), tracked by ASYNCcnt.
#define STAGE(bb, kk)                                                         \
  do {                                                                        \
    asm volatile("global_load_async_to_lds_b128 %0, %1, off"                  \
                 :: "v"((unsigned)(size_t)&As[bb][sOff]), "v"(gA + (kk))      \
                 : "memory");                                                 \
    asm volatile("global_load_async_to_lds_b128 %0, %1, off"                  \
                 :: "v"((unsigned)(size_t)&Bs[bb][sOff]), "v"(gB0 + (kk))     \
                 : "memory");                                                 \
    asm volatile("global_load_async_to_lds_b128 %0, %1, off"                  \
                 :: "v"((unsigned)(size_t)&Bs[bb][64 * 32 + sOff]),           \
                    "v"(gB1 + (kk))                                           \
                 : "memory");                                                 \
  } while (0)
#define STAGE_WAIT() asm volatile("s_wait_asynccnt 0" ::: "memory")
#else
#define STAGE(bb, kk)                                                         \
  do {                                                                        \
    *(uint4*)&As[bb][sOff]           = *(const uint4*)(gA + (kk));            \
    *(uint4*)&Bs[bb][sOff]           = *(const uint4*)(gB0 + (kk));           \
    *(uint4*)&Bs[bb][64 * 32 + sOff] = *(const uint4*)(gB1 + (kk));           \
  } while (0)
#define STAGE_WAIT() do {} while (0)
#endif

  v8f acc[4] = {{}, {}, {}, {}};

  STAGE(0, 0);
  int buf = 0;
  for (int k0 = 0; k0 < K; k0 += 32) {
    STAGE_WAIT();          // my async writes into As[buf]/Bs[buf] complete
    __syncthreads();       // everyone's writes visible; prior reads done
    if (k0 + 32 < K) STAGE(buf ^ 1, k0 + 32);  // DMA next tile during WMMA

    // A fragment: lane -> row wm*16+ln15, K = {klo..klo+7, klo+16..klo+23}
    Frag16 a;
    {
      const unsigned short* p = &As[buf][(wm * 16 + ln15) * 32 + hi * 8];
      a.q[0] = *(const uint4*)(p);
      a.q[1] = *(const uint4*)(p + 16);
    }
#pragma unroll
    for (int j = 0; j < 4; ++j) {
      // B fragment: lane -> col n, K = {klo..klo+7, klo+16..klo+23}
      Frag16 b;
      const unsigned short* p = &Bs[buf][((wn0 + j) * 16 + ln15) * 32 + hi * 8];
      b.q[0] = *(const uint4*)(p);
      b.q[1] = *(const uint4*)(p + 16);
      acc[j] = wmma_bf16(a.v, b.v, acc[j]);
    }
    buf ^= 1;
  }
#undef STAGE
#undef STAGE_WAIT

#pragma unroll
  for (int j = 0; j < 4; ++j) {
    const int col  = blockN + (wn0 + j) * 16 + ln15;
    const float bv = bias[col];
#pragma unroll
    for (int r = 0; r < 8; ++r) {
      const int row = blockM + wm * 16 + r + 8 * hi;
      float v = acc[j][r] + bv;
      if (relu) v = fmaxf(v, 0.0f);
      if (Cf) Cf[(size_t)row * N + col] = v;
      if (Ch) Ch[(size_t)row * N + col] = f2bf(v);
    }
  }
}

// ----------------------------- fused attention -----------------------------
// One wave per (batch, head). qkv: [B, S, 768] bf16 (q|k|v). out: bf16 [B,S,256].
template <int S>
__global__ __launch_bounds__(32) void attn_kernel(
    const unsigned short* __restrict__ qkv, unsigned short* __restrict__ out) {
  constexpr int KT = (S + 15) / 16;    // score tiles along keys / queries
  constexpr int KC = (KT + 1) / 2;     // 32-wide K chunks for P@V
  const int lane = threadIdx.x & 31;
  const int ln15 = lane & 15;
  const int hi   = lane >> 4;
  const int klo  = hi * 8;
  const int b    = blockIdx.x >> 3;
  const int h    = blockIdx.x & 7;
  const unsigned short* base = qkv + (size_t)b * S * 768;
  const float scale = 0.17677669529663687f;   // 1/sqrt(32)

  __shared__ unsigned short P[16 * KC * 32];  // probs, bf16, row stride KC*32

  for (int qt = 0; qt < KT; ++qt) {
    // zero P (padding keys must contribute 0 in P@V)
    for (int i = lane; i < 64 * KC; i += 32)
      ((uint4*)P)[i] = uint4{0u, 0u, 0u, 0u};
    __syncthreads();

    // Q fragment (direct bf16 load)
    Frag16 qa;
    {
      const int sq = qt * 16 + ln15;
      if (sq < S) {
        const unsigned short* qp = base + (size_t)sq * 768 + h * 32 + klo;
        qa.q[0] = *(const uint4*)(qp);
        qa.q[1] = *(const uint4*)(qp + 16);
      } else {
        qa.q[0] = uint4{0u, 0u, 0u, 0u};
        qa.q[1] = uint4{0u, 0u, 0u, 0u};
      }
    }

    // scores = (Q @ K^T) * scale
    v8f sc[KT];
#pragma unroll
    for (int kt = 0; kt < KT; ++kt) {
      Frag16 kb;
      const int sk = kt * 16 + ln15;
      if (sk < S) {
        const unsigned short* kp =
            base + (size_t)sk * 768 + 256 + h * 32 + klo;
        kb.q[0] = *(const uint4*)(kp);
        kb.q[1] = *(const uint4*)(kp + 16);
      } else {
        kb.q[0] = uint4{0u, 0u, 0u, 0u};
        kb.q[1] = uint4{0u, 0u, 0u, 0u};
      }
      v8f z = {};
      sc[kt] = wmma_bf16(qa.v, kb.v, z);
    }
#pragma unroll
    for (int kt = 0; kt < KT; ++kt) {
#pragma unroll
      for (int r = 0; r < 8; ++r) sc[kt][r] *= scale;
    }

    // mask padded key columns
#pragma unroll
    for (int kt = 0; kt < KT; ++kt) {
      if (kt * 16 + ln15 >= S) {
#pragma unroll
        for (int r = 0; r < 8; ++r) sc[kt][r] = -1e30f;
      }
    }

    // row softmax (rows live across 16-lane half-waves)
    float inv[8];
#pragma unroll
    for (int r = 0; r < 8; ++r) {
      float mx = -1e30f;
#pragma unroll
      for (int kt = 0; kt < KT; ++kt) mx = fmaxf(mx, sc[kt][r]);
#pragma unroll
      for (int off = 1; off < 16; off <<= 1)
        mx = fmaxf(mx, __shfl_xor(mx, off, 16));
      float sum = 0.0f;
#pragma unroll
      for (int kt = 0; kt < KT; ++kt) {
        float e = __expf(sc[kt][r] - mx);
        sc[kt][r] = e;
        sum += e;
      }
#pragma unroll
      for (int off = 1; off < 16; off <<= 1)
        sum += __shfl_xor(sum, off, 16);
      inv[r] = 1.0f / sum;
    }

    // P -> LDS (bf16), C-layout -> memory layout
#pragma unroll
    for (int kt = 0; kt < KT; ++kt) {
#pragma unroll
      for (int r = 0; r < 8; ++r)
        P[(r + 8 * hi) * (KC * 32) + kt * 16 + ln15] = f2bf(sc[kt][r] * inv[r]);
    }
    __syncthreads();

    // O = P @ V  (two 16-col tiles across HD=32)
    v8f o0 = {};
    v8f o1 = {};
#pragma unroll
    for (int kc = 0; kc < KC; ++kc) {
      Frag16 pa;
      const unsigned short* pp = &P[ln15 * (KC * 32) + kc * 32 + klo];
      pa.q[0] = *(const uint4*)(pp);
      pa.q[1] = *(const uint4*)(pp + 16);
#pragma unroll
      for (int nt = 0; nt < 2; ++nt) {
        Frag16 vb;
        const int col = 512 + h * 32 + nt * 16 + ln15;
        const int tk0 = kc * 32 + klo;
#pragma unroll
        for (int j = 0; j < 16; ++j) {
          const int tk = tk0 + j + (j >= 8 ? 8 : 0);
          vb.h[j] = (tk < S) ? base[(size_t)tk * 768 + col]
                             : (unsigned short)0;
        }
        if (nt == 0) o0 = wmma_bf16(pa.v, vb.v, o0);
        else         o1 = wmma_bf16(pa.v, vb.v, o1);
      }
    }

#pragma unroll
    for (int r = 0; r < 8; ++r) {
      const int sq = qt * 16 + r + 8 * hi;
      if (sq < S) {
        unsigned short* op = out + (size_t)(b * S + sq) * 256 + h * 32 + ln15;
        op[0]  = f2bf(o0[r]);
        op[16] = f2bf(o1[r]);
      }
    }
    __syncthreads();
  }
}

// ------------------------- residual + LayerNorm ----------------------------
// out[row] = LN(x[row] + h[row]) * w + b ; one wave per 256-wide row.
// Writes f32 (and optional bf16 shadow for downstream GEMMs).
__global__ __launch_bounds__(256) void add_ln_kernel(
    const float* __restrict__ x, const float* __restrict__ hbuf,
    const float* __restrict__ w, const float* __restrict__ b,
    float* __restrict__ out, unsigned short* __restrict__ outh) {
  const int lane = threadIdx.x & 31;
  const int wave = threadIdx.x >> 5;
  const size_t row = (size_t)blockIdx.x * 8 + wave;
  const float* xp = x    + row * 256 + lane * 8;
  const float* hp = hbuf + row * 256 + lane * 8;
  float4 a0 = *(const float4*)(xp);
  float4 a1 = *(const float4*)(xp + 4);
  float4 c0 = *(const float4*)(hp);
  float4 c1 = *(const float4*)(hp + 4);
  float v[8] = {a0.x + c0.x, a0.y + c0.y, a0.z + c0.z, a0.w + c0.w,
                a1.x + c1.x, a1.y + c1.y, a1.z + c1.z, a1.w + c1.w};
  float s = 0.0f, s2 = 0.0f;
#pragma unroll
  for (int r = 0; r < 8; ++r) { s += v[r]; s2 += v[r] * v[r]; }
#pragma unroll
  for (int off = 1; off < 32; off <<= 1) {
    s  += __shfl_xor(s, off, 32);
    s2 += __shfl_xor(s2, off, 32);
  }
  const float mean = s * (1.0f / 256.0f);
  const float var  = s2 * (1.0f / 256.0f) - mean * mean;
  const float rs   = rsqrtf(var + 1e-5f);
  float* op = out + row * 256 + lane * 8;
  float y[8];
#pragma unroll
  for (int r = 0; r < 8; ++r) {
    const int c = lane * 8 + r;
    y[r] = (v[r] - mean) * rs * w[c] + b[c];
    op[r] = y[r];
  }
  if (outh) {
    uint4 pk;
    pk.x = (unsigned)f2bf(y[0]) | ((unsigned)f2bf(y[1]) << 16);
    pk.y = (unsigned)f2bf(y[2]) | ((unsigned)f2bf(y[3]) << 16);
    pk.z = (unsigned)f2bf(y[4]) | ((unsigned)f2bf(y[5]) << 16);
    pk.w = (unsigned)f2bf(y[6]) | ((unsigned)f2bf(y[7]) << 16);
    *(uint4*)(outh + row * 256 + lane * 8) = pk;
  }
}

// ------------------------------ CLS gather ---------------------------------
// XT[seg*64 + t, d] = Xs[(t*256+seg)*9 + 0, d]  (f32 master + bf16 shadow)
__global__ __launch_bounds__(256) void gather_cls(
    const float* __restrict__ Xs, float* __restrict__ XT,
    unsigned short* __restrict__ XTh) {
  const int r = blockIdx.x;      // seg*64 + t
  const int d = threadIdx.x;
  const int seg = r >> 6;
  const int t   = r & 63;
  const float v = Xs[(size_t)(t * 256 + seg) * 9 * 256 + d];
  XT[(size_t)r * 256 + d]  = v;
  XTh[(size_t)r * 256 + d] = f2bf(v);
}

// ---------------------------------------------------------------------------
extern "C" void kernel_launch(void* const* d_in, const int* in_sizes, int n_in,
                              void* d_out, int out_size, void* d_ws, size_t ws_size,
                              hipStream_t stream) {
  (void)in_sizes; (void)n_in; (void)out_size; (void)ws_size;

  const float* mv     = (const float*)d_in[0];
  const int*   sse    = (const int*)d_in[1];
  const float* w_in   = (const float*)d_in[2];
  const float* b_in   = (const float*)d_in[3];
  const float* s_qkvw = (const float*)d_in[4];
  const float* s_qkvb = (const float*)d_in[5];
  const float* s_ow   = (const float*)d_in[6];
  const float* s_ob   = (const float*)d_in[7];
  const float* s_l1w  = (const float*)d_in[8];
  const float* s_l1b  = (const float*)d_in[9];
  const float* s_l2w  = (const float*)d_in[10];
  const float* s_l2b  = (const float*)d_in[11];
  const float* s_ln1w = (const float*)d_in[12];
  const float* s_ln1b = (const float*)d_in[13];
  const float* s_ln2w = (const float*)d_in[14];
  const float* s_ln2b = (const float*)d_in[15];
  const float* t_qkvw = (const float*)d_in[16];
  const float* t_qkvb = (const float*)d_in[17];
  const float* t_ow   = (const float*)d_in[18];
  const float* t_ob   = (const float*)d_in[19];
  const float* t_l1w  = (const float*)d_in[20];
  const float* t_l1b  = (const float*)d_in[21];
  const float* t_l2w  = (const float*)d_in[22];
  const float* t_l2b  = (const float*)d_in[23];
  const float* t_ln1w = (const float*)d_in[24];
  const float* t_ln1b = (const float*)d_in[25];
  const float* t_ln2w = (const float*)d_in[26];
  const float* t_ln2b = (const float*)d_in[27];

  char* ws = (char*)d_ws;
  size_t off = 0;
  auto take = [&](size_t bytes) -> char* {
    char* p = ws + off;
    off += (bytes + 255) & ~(size_t)255;
    return p;
  };

  // bf16 weights
  unsigned short* wq_s  = (unsigned short*)take((size_t)2 * 768 * 256 * 2);
  unsigned short* wo_s  = (unsigned short*)take((size_t)2 * 256 * 256 * 2);
  unsigned short* wl1_s = (unsigned short*)take((size_t)2 * 128 * 256 * 2);
  unsigned short* wl2_s = (unsigned short*)take((size_t)2 * 256 * 128 * 2);
  unsigned short* wq_t  = (unsigned short*)take((size_t)2 * 768 * 256 * 2);
  unsigned short* wo_t  = (unsigned short*)take((size_t)2 * 256 * 256 * 2);
  unsigned short* wl1_t = (unsigned short*)take((size_t)2 * 128 * 256 * 2);
  unsigned short* wl2_t = (unsigned short*)take((size_t)2 * 256 * 128 * 2);

  // activations
  const int Ms = 147456;   // 16384 * 9
  const int Mt = 16384;    // 256 * 64
  float*          Xs   = (float*)take((size_t)Ms * 256 * 4);          // f32 master
  unsigned short* Xsh  = (unsigned short*)take((size_t)Ms * 256 * 2); // bf16 shadow
  unsigned short* QKVh = (unsigned short*)take((size_t)Ms * 768 * 2);
  unsigned short* ATh  = (unsigned short*)take((size_t)Ms * 256 * 2);
  float*          Hb   = (float*)take((size_t)Ms * 256 * 4);
  unsigned short* Fh   = (unsigned short*)take((size_t)Ms * 128 * 2);
  float*          XT   = (float*)take((size_t)Mt * 256 * 4);
  unsigned short* XTh  = (unsigned short*)take((size_t)Mt * 256 * 2);

  auto cvt = [&](const float* src, unsigned short* dst, int n) {
    cvt_kernel<<<(n + 255) / 256, 256, 0, stream>>>(src, dst, n);
  };
  cvt(s_qkvw, wq_s,  2 * 768 * 256);
  cvt(s_ow,   wo_s,  2 * 256 * 256);
  cvt(s_l1w,  wl1_s, 2 * 128 * 256);
  cvt(s_l2w,  wl2_s, 2 * 256 * 128);
  cvt(t_qkvw, wq_t,  2 * 768 * 256);
  cvt(t_ow,   wo_t,  2 * 256 * 256);
  cvt(t_l1w,  wl1_t, 2 * 128 * 256);
  cvt(t_l2w,  wl2_t, 2 * 256 * 128);

  embed_kernel<<<Ms, 256, 0, stream>>>(mv, sse, w_in, b_in, Xs, Xsh);

  // ---- spatial encoder (B=16384, S=9) ----
  for (int l = 0; l < 2; ++l) {
    gemm_bf16<<<dim3(Ms / 64, 768 / 128), 256, 0, stream>>>(
        Xsh, wq_s + (size_t)l * 768 * 256, s_qkvb + l * 768,
        nullptr, QKVh, Ms, 768, 256, 0);
    attn_kernel<9><<<16384 * 8, 32, 0, stream>>>(QKVh, ATh);
    gemm_bf16<<<dim3(Ms / 64, 256 / 128), 256, 0, stream>>>(
        ATh, wo_s + (size_t)l * 256 * 256, s_ob + l * 256,
        Hb, nullptr, Ms, 256, 256, 0);
    add_ln_kernel<<<Ms / 8, 256, 0, stream>>>(
        Xs, Hb, s_ln1w + l * 256, s_ln1b + l * 256, Xs, Xsh);
    gemm_bf16<<<dim3(Ms / 64, 128 / 128), 256, 0, stream>>>(
        Xsh, wl1_s + (size_t)l * 128 * 256, s_l1b + l * 128,
        nullptr, Fh, Ms, 128, 256, 1);
    gemm_bf16<<<dim3(Ms / 64, 256 / 128), 256, 0, stream>>>(
        Fh, wl2_s + (size_t)l * 256 * 128, s_l2b + l * 256,
        Hb, nullptr, Ms, 256, 128, 0);
    add_ln_kernel<<<Ms / 8, 256, 0, stream>>>(
        Xs, Hb, s_ln2w + l * 256, s_ln2b + l * 256, Xs, Xsh);
  }

  gather_cls<<<Mt, 256, 0, stream>>>(Xs, XT, XTh);

  // ---- temporal encoder (B=256, S=64) ----
  float* outF = (float*)d_out;
  for (int l = 0; l < 2; ++l) {
    gemm_bf16<<<dim3(Mt / 64, 768 / 128), 256, 0, stream>>>(
        XTh, wq_t + (size_t)l * 768 * 256, t_qkvb + l * 768,
        nullptr, QKVh, Mt, 768, 256, 0);
    attn_kernel<64><<<256 * 8, 32, 0, stream>>>(QKVh, ATh);
    gemm_bf16<<<dim3(Mt / 64, 256 / 128), 256, 0, stream>>>(
        ATh, wo_t + (size_t)l * 256 * 256, t_ob + l * 256,
        Hb, nullptr, Mt, 256, 256, 0);
    add_ln_kernel<<<Mt / 8, 256, 0, stream>>>(
        XT, Hb, t_ln1w + l * 256, t_ln1b + l * 256, XT, XTh);
    gemm_bf16<<<dim3(Mt / 64, 128 / 128), 256, 0, stream>>>(
        XTh, wl1_t + (size_t)l * 128 * 256, t_l1b + l * 128,
        nullptr, Fh, Mt, 128, 256, 1);
    gemm_bf16<<<dim3(Mt / 64, 256 / 128), 256, 0, stream>>>(
        Fh, wl2_t + (size_t)l * 256 * 128, t_l2b + l * 256,
        Hb, nullptr, Mt, 256, 128, 0);
    add_ln_kernel<<<Mt / 8, 256, 0, stream>>>(
        XT, Hb, t_ln2w + l * 256, t_ln2b + l * 256,
        (l == 1) ? outF : XT, (l == 1) ? nullptr : XTh);
  }
}
